// WindowTransformerBlock_283467842427
// MI455X (gfx1250) — matrix operand order repaired and checked by reading, not compiled, not run on hardware
//
#include <hip/hip_runtime.h>
#include <hip/hip_bf16.h>

typedef __attribute__((ext_vector_type(16))) __bf16 v16bf;
typedef __attribute__((ext_vector_type(8)))  __bf16 v8bf;
typedef __attribute__((ext_vector_type(8)))  float  v8f;

// Problem constants
#define BB 64
#define HH 56
#define CCH 256
#define WIN 7
#define SHIFT 3
#define HEADS 8
#define HD 32
#define NTOK 49
#define NPAD 64
#define NW 64                      // windows per image
#define NWIN (BB * NW)             // 4096 windows total
#define MPAD ((size_t)NWIN * NPAD) // 262144 padded rows
#define MTOK ((size_t)BB * HH * HH)// 200704 real tokens
#define KCHUNK 128                 // K elements staged per LDS chunk
#define LDS_STRIDE 136             // 128 + 8 pad (bank spread)
static constexpr float SCALE_F = 0.17677669529663689f; // 32^-0.5

// ---------------------------------------------------------------- helpers
static __device__ __forceinline__ v16bf load_a16(const __bf16* row, int k0, int half) {
  // 16-bit A fragment (16x32): lane m=lane%16 holds k in [h*16,h*16+16)
  v8bf lo = *reinterpret_cast<const v8bf*>(row + k0 + half * 16);
  v8bf hi = *reinterpret_cast<const v8bf*>(row + k0 + half * 16 + 8);
  v16bf a;
#pragma unroll
  for (int i = 0; i < 8; ++i) { a[i] = lo[i]; a[i + 8] = hi[i]; }
  return a;
}

// Cooperative async copy of a 64 x 128 bf16 weight chunk into LDS.
// 256 threads, each issues 4 global_load_async_to_lds_b128 (16B per lane).
static __device__ __forceinline__ void async_load_chunk(const __bf16* __restrict__ W, int ldw,
                                                        int n0, int kbase,
                                                        unsigned ldsBufByte, int tid) {
  const int row = tid >> 2;   // 0..63
  const int seg = tid & 3;    // 0..3 : 32 elements each
#pragma unroll
  for (int i = 0; i < 4; ++i) {
    const __bf16* gp = W + (size_t)(n0 + row) * ldw + kbase + seg * 32 + i * 8;
    unsigned lp = ldsBufByte + (unsigned)((row * LDS_STRIDE + seg * 32 + i * 8) * 2);
    asm volatile("global_load_async_to_lds_b128 %0, %1, off"
                 :: "v"(lp), "v"(gp) : "memory");
  }
}

// Block-cooperative GEMM: 8 waves x (16x64 tile), B strip staged in LDS,
// double-buffered async chunks of 64x128 bf16.
static __device__ __forceinline__ void gemm_block(const __bf16* __restrict__ A, int lda,
                                                  const __bf16* __restrict__ W, int ldw,
                                                  int K, int m0, int n0,
                                                  int lane, int tid,
                                                  __bf16* ldsB, v8f c[4]) {
  const unsigned ldsByte = (unsigned)(uintptr_t)ldsB;
  const int half = lane >> 4, lm = lane & 15;
  const __bf16* arow = A + (size_t)(m0 + lm) * lda;
  const int nch = K / KCHUNK;
  async_load_chunk(W, ldw, n0, 0, ldsByte, tid);
  asm volatile("s_wait_asynccnt 0x0" ::: "memory");
  __syncthreads();
  for (int ch = 0; ch < nch; ++ch) {
    if (ch + 1 < nch)
      async_load_chunk(W, ldw, n0, (ch + 1) * KCHUNK,
                       ldsByte + (unsigned)(((ch + 1) & 1) * (64 * LDS_STRIDE * 2)), tid);
    const __bf16* bbuf = ldsB + (ch & 1) * (64 * LDS_STRIDE);
    // Prefetch all four A sub-fragments of this chunk (global, 2x b128 each).
    v16bf a[4];
#pragma unroll
    for (int ks = 0; ks < 4; ++ks) a[ks] = load_a16(arow, ch * KCHUNK + ks * 32, half);
#pragma unroll
    for (int ks = 0; ks < 4; ++ks) {
      // Gather all four B fragments first (one ds clause + single wait),
      // then issue the four WMMAs back-to-back.
      v16bf b[4];
#pragma unroll
      for (int j = 0; j < 4; ++j)
        b[j] = *reinterpret_cast<const v16bf*>(bbuf + (j * 16 + lm) * LDS_STRIDE +
                                               ks * 32 + half * 16);
#pragma unroll
      for (int j = 0; j < 4; ++j)
        c[j] = __builtin_amdgcn_wmma_f32_16x16x32_bf16(false, a[ks], false, b[j],
                                                       (short)0, c[j], false, false);
    }
    asm volatile("s_wait_asynccnt 0x0" ::: "memory");
    __syncthreads();
  }
}

static __device__ __forceinline__ int regf(int p) { return (p < 49) ? 0 : ((p < 53) ? 1 : 2); }

// ---------------------------------------------------------------- kernels
__global__ void cvt_bf16(const float* __restrict__ src, __bf16* __restrict__ dst, int n) {
  int i = blockIdx.x * 256 + threadIdx.x;
  if (i < n) dst[i] = (__bf16)src[i];
}

// LN1 + roll(-3,-3) + window partition -> padded [4096][64][256] bf16
__global__ void ln1_window(const float* __restrict__ x, const float* __restrict__ w,
                           const float* __restrict__ b, __bf16* __restrict__ Hwin) {
  __shared__ float red[256];
  __shared__ float red2[256];
  const int p = blockIdx.x, tid = threadIdx.x;
  const int win = p >> 6, tok = p & 63;
  const size_t orow = (size_t)p * CCH;
  if (tok >= NTOK) { Hwin[orow + tid] = (__bf16)0.0f; return; }
  const int bb = win >> 6, wi = win & 63, wh = wi >> 3, ww = wi & 7;
  const int i = tok / 7, j = tok % 7;
  int hh = wh * 7 + i + SHIFT; if (hh >= HH) hh -= HH;
  int wc = ww * 7 + j + SHIFT; if (wc >= HH) wc -= HH;
  const float* src = x + (((size_t)bb * (HH * HH)) + hh * HH + wc) * CCH;
  float v = src[tid];
  red[tid] = v; red2[tid] = v * v;
  __syncthreads();
  for (int s = 128; s > 0; s >>= 1) {
    if (tid < s) { red[tid] += red[tid + s]; red2[tid] += red2[tid + s]; }
    __syncthreads();
  }
  float mean = red[0] * (1.0f / 256.0f);
  float var  = red2[0] * (1.0f / 256.0f) - mean * mean;
  float o = (v - mean) * rsqrtf(var + 1e-5f) * w[tid] + b[tid];
  Hwin[orow + tid] = (__bf16)o;
}

// LN2 over X1 -> bf16
__global__ void ln2_rows(const float* __restrict__ X1, const float* __restrict__ w,
                         const float* __restrict__ b, __bf16* __restrict__ H2) {
  __shared__ float red[256];
  __shared__ float red2[256];
  const int tid = threadIdx.x;
  const size_t row = (size_t)blockIdx.x * CCH;
  float v = X1[row + tid];
  red[tid] = v; red2[tid] = v * v;
  __syncthreads();
  for (int s = 128; s > 0; s >>= 1) {
    if (tid < s) { red[tid] += red[tid + s]; red2[tid] += red2[tid + s]; }
    __syncthreads();
  }
  float mean = red[0] * (1.0f / 256.0f);
  float var  = red2[0] * (1.0f / 256.0f) - mean * mean;
  H2[row + tid] = (__bf16)((v - mean) * rsqrtf(var + 1e-5f) * w[tid] + b[tid]);
}

// QKV GEMM: [262144 x 256] x [768 x 256]^T ; scatter to Q, K, V^T (bf16)
__global__ void qkv_gemm(const __bf16* __restrict__ Hwin, const __bf16* __restrict__ Wq,
                         const float* __restrict__ qkv_b, __bf16* __restrict__ Q,
                         __bf16* __restrict__ Kb, __bf16* __restrict__ Vt) {
  __shared__ __bf16 ldsB[2 * 64 * LDS_STRIDE];
  const int units_n = 768 / 64;
  const int lane = threadIdx.x & 31, wave = threadIdx.x >> 5;
  const int m0 = (blockIdx.x / units_n) * 128 + wave * 16;
  const int n0 = (blockIdx.x % units_n) * 64;
  v8f c[4] = {};
  gemm_block(Hwin, CCH, Wq, CCH, CCH, m0, n0, lane, threadIdx.x, ldsB, c);
  const int half = lane >> 4, lm = lane & 15;
#pragma unroll
  for (int j = 0; j < 4; ++j) {
    const int n = n0 + j * 16 + lm;
    const int sec = n >> 8, f = n & 255, head = f >> 5, d = f & 31;
    const float bias = qkv_b[n];
#pragma unroll
    for (int r = 0; r < 8; ++r) {
      const int m = m0 + r + half * 8;
      const int win = m >> 6, row = m & 63;
      const size_t bh = (size_t)win * HEADS + head;
      float v = c[j][r] + bias;
      if (sec == 0)      Q [(bh * NPAD + row) * HD + d] = (__bf16)(v * SCALE_F);
      else if (sec == 1) Kb[(bh * NPAD + row) * HD + d] = (__bf16)v;
      else               Vt[(bh * HD + d) * NPAD + row] = (__bf16)v; // transposed
    }
  }
}

// Attention: one block (4 waves) per (window, head)
__global__ void attn_kernel(const __bf16* __restrict__ Q, const __bf16* __restrict__ Kb,
                            const __bf16* __restrict__ Vt, const float* __restrict__ relTab,
                            __bf16* __restrict__ AO) {
  __shared__ float  S[64][65];
  __shared__ __bf16 P[64][72];
  const int blk = blockIdx.x;
  const int win = blk >> 3, head = blk & 7;
  const int tid = threadIdx.x, lane = tid & 31, wave = tid >> 5;
  const int half = lane >> 4, lm = lane & 15;
  const int m0 = wave * 16;
  const size_t bh = (size_t)win * HEADS + head;
  const __bf16* qb = Q  + bh * NPAD * HD;
  const __bf16* kb = Kb + bh * NPAD * HD;
  const int wi = win & 63, wh = wi >> 3, ww = wi & 7;

  // ---- Phase 1: S = q * k^T (+bias +mask) into LDS
  v16bf a = load_a16(qb + (size_t)(m0 + lm) * HD, 0, half);
#pragma unroll
  for (int j = 0; j < 4; ++j) {
    v16bf b = *reinterpret_cast<const v16bf*>(kb + (size_t)(j * 16 + lm) * HD + half * 16);
    v8f c = {};
    c = __builtin_amdgcn_wmma_f32_16x16x32_bf16(false, a, false, b, (short)0, c, false, false);
    const int col = j * 16 + lm;
    const int ck = (col < NTOK) ? col : 48;
    const int ik = ck / 7, jk = ck % 7;
    const int regK = regf(wh * 7 + ik) * 3 + regf(ww * 7 + jk);
#pragma unroll
    for (int r = 0; r < 8; ++r) {
      const int m = m0 + r + half * 8;
      const int mq = (m < NTOK) ? m : 48;
      const int iq = mq / 7, jq = mq % 7;
      const float bias = relTab[((iq - ik + 6) * 13 + (jq - jk + 6)) * HEADS + head];
      const int regQ = regf(wh * 7 + iq) * 3 + regf(ww * 7 + jq);
      float val = c[r] + bias + ((regQ != regK) ? -100.0f : 0.0f);
      if (col >= NTOK) val = -1e30f;
      S[m][col] = val;
    }
  }
  __syncthreads();

  // ---- Phase 2: row softmax -> P (bf16)
  if (tid < 64) {
    float mx = -1e30f;
    for (int n = 0; n < 64; ++n) mx = fmaxf(mx, S[tid][n]);
    float sum = 0.0f;
    for (int n = 0; n < 64; ++n) { float e = __expf(S[tid][n] - mx); S[tid][n] = e; sum += e; }
    const float inv = 1.0f / fmaxf(sum, 1e-20f);
    for (int n = 0; n < 64; ++n) P[tid][n] = (__bf16)(S[tid][n] * inv);
    for (int n = 64; n < 72; ++n) P[tid][n] = (__bf16)0.0f;
  }
  __syncthreads();

  // ---- Phase 3: O = P * V   (V stored transposed: [HD][NPAD])
  const __bf16* vb = Vt + bh * HD * NPAD;
#pragma unroll
  for (int j = 0; j < 2; ++j) {
    v8f c = {};
#pragma unroll
    for (int kk = 0; kk < 2; ++kk) {
      const int k0 = kk * 32;
      v16bf a2 = load_a16(&P[m0 + lm][0], k0, half);
      v16bf b2 = *reinterpret_cast<const v16bf*>(vb + (size_t)(j * 16 + lm) * NPAD + k0 + half * 16);
      c = __builtin_amdgcn_wmma_f32_16x16x32_bf16(false, a2, false, b2, (short)0, c, false, false);
    }
#pragma unroll
    for (int r = 0; r < 8; ++r) {
      const int m = m0 + r + half * 8;
      AO[((size_t)win * NPAD + m) * CCH + head * HD + j * 16 + lm] = (__bf16)c[r];
    }
  }
}

// Proj GEMM + window reverse + roll(+3,+3) + residual -> X1 (f32)
__global__ void proj_gemm(const __bf16* __restrict__ AO, const __bf16* __restrict__ Wp,
                          const float* __restrict__ proj_b, const float* __restrict__ x,
                          float* __restrict__ X1) {
  __shared__ __bf16 ldsB[2 * 64 * LDS_STRIDE];
  const int units_n = 256 / 64;
  const int lane = threadIdx.x & 31, wave = threadIdx.x >> 5;
  const int m0 = (blockIdx.x / units_n) * 128 + wave * 16;
  const int n0 = (blockIdx.x % units_n) * 64;
  v8f c[4] = {};
  gemm_block(AO, CCH, Wp, CCH, CCH, m0, n0, lane, threadIdx.x, ldsB, c);
  const int half = lane >> 4, lm = lane & 15;
#pragma unroll
  for (int j = 0; j < 4; ++j) {
    const int n = n0 + j * 16 + lm;
    const float bias = proj_b[n];
#pragma unroll
    for (int r = 0; r < 8; ++r) {
      const int m = m0 + r + half * 8;
      const int win = m >> 6, tok = m & 63;
      if (tok < NTOK) {
        const int bb = win >> 6, wi = win & 63, wh = wi >> 3, ww = wi & 7;
        int hh = wh * 7 + tok / 7 + SHIFT; if (hh >= HH) hh -= HH;
        int wc = ww * 7 + tok % 7 + SHIFT; if (wc >= HH) wc -= HH;
        const size_t idx = (((size_t)bb * (HH * HH)) + hh * HH + wc) * CCH + n;
        X1[idx] = x[idx] + c[j][r] + bias;
      }
    }
  }
}

// fc1 GEMM + exact GELU -> bf16
__global__ void fc1_gemm(const __bf16* __restrict__ H2, const __bf16* __restrict__ W1,
                         const float* __restrict__ b1, __bf16* __restrict__ G) {
  __shared__ __bf16 ldsB[2 * 64 * LDS_STRIDE];
  const int units_n = 1024 / 64;
  const int lane = threadIdx.x & 31, wave = threadIdx.x >> 5;
  const int m0 = (blockIdx.x / units_n) * 128 + wave * 16;
  const int n0 = (blockIdx.x % units_n) * 64;
  v8f c[4] = {};
  gemm_block(H2, CCH, W1, CCH, CCH, m0, n0, lane, threadIdx.x, ldsB, c);
  const int half = lane >> 4, lm = lane & 15;
#pragma unroll
  for (int j = 0; j < 4; ++j) {
    const int n = n0 + j * 16 + lm;
    const float bias = b1[n];
#pragma unroll
    for (int r = 0; r < 8; ++r) {
      const int m = m0 + r + half * 8;
      float v = c[j][r] + bias;
      float g = 0.5f * v * (1.0f + erff(v * 0.7071067811865476f));
      G[(size_t)m * 1024 + n] = (__bf16)g;
    }
  }
}

// fc2 GEMM + residual -> d_out (f32)
__global__ void fc2_gemm(const __bf16* __restrict__ G, const __bf16* __restrict__ W2,
                         const float* __restrict__ b2, const float* __restrict__ X1,
                         float* __restrict__ out) {
  __shared__ __bf16 ldsB[2 * 64 * LDS_STRIDE];
  const int units_n = 256 / 64;
  const int lane = threadIdx.x & 31, wave = threadIdx.x >> 5;
  const int m0 = (blockIdx.x / units_n) * 128 + wave * 16;
  const int n0 = (blockIdx.x % units_n) * 64;
  v8f c[4] = {};
  gemm_block(G, 1024, W2, 1024, 1024, m0, n0, lane, threadIdx.x, ldsB, c);
  const int half = lane >> 4, lm = lane & 15;
#pragma unroll
  for (int j = 0; j < 4; ++j) {
    const int n = n0 + j * 16 + lm;
    const float bias = b2[n];
#pragma unroll
    for (int r = 0; r < 8; ++r) {
      const int m = m0 + r + half * 8;
      const size_t idx = (size_t)m * CCH + n;
      out[idx] = X1[idx] + c[j][r] + bias;
    }
  }
}

// ---------------------------------------------------------------- launch
extern "C" void kernel_launch(void* const* d_in, const int* in_sizes, int n_in,
                              void* d_out, int out_size, void* d_ws, size_t ws_size,
                              hipStream_t stream) {
  (void)in_sizes; (void)n_in; (void)out_size; (void)ws_size;
  const float* x      = (const float*)d_in[0];
  const float* n1w    = (const float*)d_in[1];
  const float* n1b    = (const float*)d_in[2];
  const float* qkvw   = (const float*)d_in[3];
  const float* qkvb   = (const float*)d_in[4];
  const float* projw  = (const float*)d_in[5];
  const float* projb  = (const float*)d_in[6];
  const float* relt   = (const float*)d_in[7];
  const float* n2w    = (const float*)d_in[8];
  const float* n2b    = (const float*)d_in[9];
  const float* fc1w   = (const float*)d_in[10];
  const float* fc1b   = (const float*)d_in[11];
  const float* fc2w   = (const float*)d_in[12];
  const float* fc2b   = (const float*)d_in[13];

  char* ws = (char*)d_ws;
  size_t o = 0;
  auto alloc = [&](size_t bytes) { size_t cur = o; o += (bytes + 255) & ~(size_t)255; return cur; };

  const size_t oWq = alloc(768 * 256 * 2);
  const size_t oWp = alloc(256 * 256 * 2);
  const size_t oW1 = alloc(1024 * 256 * 2);
  const size_t oW2 = alloc(256 * 1024 * 2);
  const size_t oHwin = alloc(MPAD * CCH * 2);            // reused later for H2
  const size_t oQ  = alloc(MPAD * CCH * 2);              // reused later for G (fc1 out)
  const size_t oK  = alloc(MPAD * CCH * 2);
  const size_t oV  = alloc(MPAD * CCH * 2);
  const size_t oAO = alloc(MPAD * CCH * 2);
  const size_t oX1 = alloc(MTOK * CCH * 4);

  __bf16* Wq = (__bf16*)(ws + oWq);
  __bf16* Wp = (__bf16*)(ws + oWp);
  __bf16* W1 = (__bf16*)(ws + oW1);
  __bf16* W2 = (__bf16*)(ws + oW2);
  __bf16* Hwin = (__bf16*)(ws + oHwin);
  __bf16* Qb = (__bf16*)(ws + oQ);
  __bf16* Kb = (__bf16*)(ws + oK);
  __bf16* Vt = (__bf16*)(ws + oV);
  __bf16* AO = (__bf16*)(ws + oAO);
  float*  X1 = (float*)(ws + oX1);
  __bf16* H2 = Hwin;                 // Hwin dead after qkv_gemm
  __bf16* G  = Qb;                   // Q/K/V/AO dead after proj_gemm (411MB fits in 536MB)
  float*  out = (float*)d_out;

  // 1) weights -> bf16
  cvt_bf16<<<(768 * 256 + 255) / 256, 256, 0, stream>>>(qkvw, Wq, 768 * 256);
  cvt_bf16<<<(256 * 256 + 255) / 256, 256, 0, stream>>>(projw, Wp, 256 * 256);
  cvt_bf16<<<(1024 * 256 + 255) / 256, 256, 0, stream>>>(fc1w, W1, 1024 * 256);
  cvt_bf16<<<(256 * 1024 + 255) / 256, 256, 0, stream>>>(fc2w, W2, 256 * 1024);

  // 2) LN1 + shift + window partition (padded)
  ln1_window<<<(int)MPAD, 256, 0, stream>>>(x, n1w, n1b, Hwin);

  // 3) QKV GEMM: (262144/128)*(768/64) blocks
  qkv_gemm<<<24576, 256, 0, stream>>>(Hwin, Wq, qkvb, Qb, Kb, Vt);

  // 4) attention: one block per (window, head)
  attn_kernel<<<NWIN * HEADS, 128, 0, stream>>>(Qb, Kb, Vt, relt, AO);

  // 5) proj + reverse/unshift + residual: (262144/128)*(256/64)
  proj_gemm<<<8192, 256, 0, stream>>>(AO, Wp, projb, x, X1);

  // 6) LN2
  ln2_rows<<<(int)MTOK, 256, 0, stream>>>(X1, n2w, n2b, H2);

  // 7) fc1 + GELU: (200704/128)*(1024/64)
  fc1_gemm<<<25088, 256, 0, stream>>>(H2, W1, fc1b, G);

  // 8) fc2 + residual -> out: (200704/128)*(256/64)
  fc2_gemm<<<6272, 256, 0, stream>>>(G, W2, fc2b, X1, out);
}
